// GCN_35527969472874
// MI455X (gfx1250) — matrix operand order repaired and checked by reading, not compiled, not run on hardware
//
#include <hip/hip_runtime.h>

// ---------------------------------------------------------------------------
// Sizes (static, from the reference)
// ---------------------------------------------------------------------------
#define N0v 100000
#define N1v 25000
#define N2v 6250
#define E0v 1600000
#define E1v 400000
#define E2v 100000
#define NFv 128
#define NHv 128
#define NCv 40

typedef __attribute__((ext_vector_type(16))) __bf16 v16bf;
typedef __attribute__((ext_vector_type(8)))  float  v8f;
typedef __attribute__((ext_vector_type(4)))  unsigned int v4u;
typedef __attribute__((ext_vector_type(8)))  int v8i_;
typedef __attribute__((ext_vector_type(4)))  int v4i_;

union FragBF {
    v16bf v;
    unsigned short u[16];
    uint4 q[2];
};

// bf16 round-to-nearest-even split helpers (manual, avoids scalar __bf16 math)
__device__ __forceinline__ unsigned short f2bf_rne(float f) {
    unsigned u = __float_as_uint(f);
    unsigned r = u + 0x7FFFu + ((u >> 16) & 1u);
    return (unsigned short)(r >> 16);
}
__device__ __forceinline__ float bf2f(unsigned short h) {
    return __uint_as_float(((unsigned)h) << 16);
}

// order-preserving float <-> uint map for atomicMax-based segment max
__device__ __forceinline__ unsigned f2ord(float f) {
    unsigned b = __float_as_uint(f);
    return (b & 0x80000000u) ? ~b : (b | 0x80000000u);
}
__device__ __forceinline__ float ord2f(unsigned u) {
    return (u & 0x80000000u) ? __uint_as_float(u ^ 0x80000000u) : __uint_as_float(~u);
}

// ---------------------------------------------------------------------------
// fp32 -> (bf16 hi, bf16 lo) split (for the large activation matrices)
// ---------------------------------------------------------------------------
__global__ void k_split_bf16(const float* __restrict__ x,
                             unsigned short* __restrict__ hi,
                             unsigned short* __restrict__ lo, int n) {
    int i = blockIdx.x * blockDim.x + threadIdx.x;
    if (i >= n) return;
    float f = x[i];
    unsigned short h = f2bf_rne(f);
    hi[i] = h;
    lo[i] = f2bf_rne(f - bf2f(h));
}

__global__ void k_zfill(float* __restrict__ p, int n) {
    int i = blockIdx.x * blockDim.x + threadIdx.x;
    if (i < n) p[i] = 0.0f;
}

// ---------------------------------------------------------------------------
// Pack weight matrix W[K,N] (fp32) into WMMA B-fragment order, bf16 hi/lo.
// Element t = (((sel*ksteps + ks)*tiles_n) + tn)*512 + lane*16 + j
//   holds (sel ? lo : hi) of W[ks*32 + ((lane&16)?8:0) + (j&7) + ((j&8)?16:0)][tn*16 + (lane&15)]
// so a wave can read one whole fragment as 2 contiguous 128-bit loads per lane.
// ---------------------------------------------------------------------------
__global__ void k_pack_b(const float* __restrict__ W, unsigned short* __restrict__ Bpk,
                         int N, int K) {
    const int tiles_n = (N + 15) >> 4;
    const int ksteps  = K >> 5;
    const int total   = 2 * ksteps * tiles_n * 512;
    int t = blockIdx.x * blockDim.x + threadIdx.x;
    if (t >= total) return;
    int j    = t & 15;
    int lane = (t >> 4) & 31;
    int rem  = t >> 9;
    int tn   = rem % tiles_n; rem /= tiles_n;
    int ks   = rem % ksteps;
    int sel  = rem / ksteps;
    int n = tn * 16 + (lane & 15);
    int k = ks * 32 + ((lane & 16) ? 8 : 0) + (j & 7) + ((j & 8) ? 16 : 0);
    unsigned short v = 0;
    if (n < N) {
        float f = W[(size_t)k * N + n];
        unsigned short h = f2bf_rne(f);
        v = sel ? f2bf_rne(f - bf2f(h)) : h;
    }
    Bpk[t] = v;
}

// ---------------------------------------------------------------------------
// WMMA GEMM:  C[M,N] = A[M,K] @ B[K,N]   (bf16x3 emulated fp32)
// One workgroup (8 waves) per 16-row M-tile; wave w owns N-tile w.
// Packed B fragments are DMA'd once per workgroup into LDS by the Tensor Data
// Mover (tensor_load_to_lds + s_wait_tensorcnt), then consumed as contiguous
// 128-bit ds loads. 3 x v_wmma_f32_16x16x32_bf16 per 32-wide K step.
// Requires K <= 128 (mult of 32), N <= 128.
// ---------------------------------------------------------------------------
__global__ void __launch_bounds__(256) k_wmma_gemm_bf16x3(
    const unsigned short* __restrict__ Ahi,
    const unsigned short* __restrict__ Alo,
    const unsigned short* __restrict__ Bpk,
    float* __restrict__ C, int M, int N, int K) {
    __shared__ unsigned short bsh[2 * 4 * 8 * 512];   // 64 KB (of 320 KB/WGP)

    const int tiles_n = (N + 15) >> 4;
    const int ksteps  = K >> 5;
    const int nWords  = (2 * ksteps * tiles_n * 512) >> 1;   // dwords of packed B

    // ---- TDM: async-DMA the packed weights into LDS (wave 0 issues) ----
    if (threadIdx.x == 0) {
        unsigned long long gaddr = (unsigned long long)(size_t)Bpk;
        unsigned lds_addr = (unsigned)(size_t)(&bsh[0]);   // low 32 bits = LDS offset
        v4u g0;
        g0[0] = 1u;                                        // count=1, user descriptor
        g0[1] = lds_addr;                                  // D#.lds_addr
        g0[2] = (unsigned)gaddr;                           // D#.global_addr[31:0]
        g0[3] = (unsigned)((gaddr >> 32) & 0x1FFFFFFu)     // D#.global_addr[56:32]
                | 0x80000000u;                             // type=2 ("image")
        v8i_ g1;
        g1[0] = (int)(2u << 16);                           // data_size=4B, no mask
        g1[1] = (int)((nWords & 0xFFFF) << 16);            // tensor_dim0[15:0]
        g1[2] = (int)(((unsigned)nWords >> 16) | (1u << 16)); // dim0 hi | tensor_dim1=1
        g1[3] = (int)((nWords & 0xFFFF) << 16);            // tile_dim0 = nWords
        g1[4] = 1;                                         // tile_dim1 = 1
        g1[5] = nWords;                                    // tensor_dim0_stride
        g1[6] = 0;
        g1[7] = 0;
        v4i_ gz4 = {0, 0, 0, 0};                           // groups 2/3 unused (<=2D)
        v8i_ gz8 = {0, 0, 0, 0, 0, 0, 0, 0};               // trailing group (clang-23 arity)
        __builtin_amdgcn_tensor_load_to_lds(g0, g1, gz4, gz4, gz8, 0);
        __builtin_amdgcn_s_wait_tensorcnt(0);
    }
    __syncthreads();

    const int lane = threadIdx.x & 31;
    const int wave = threadIdx.x >> 5;
    if (wave >= tiles_n) return;                           // wave-uniform

    const int row0 = blockIdx.x << 4;
    const int am = row0 + (lane & 15);
    const bool arow_ok = (am < M);
    const int aoff = (lane & 16) ? 8 : 0;                  // ISA 16-bit A layout

    v8f acc = {0.f, 0.f, 0.f, 0.f, 0.f, 0.f, 0.f, 0.f};

    for (int ks = 0; ks < ksteps; ++ks) {
        FragBF ah, al, bh, bl;
        if (arow_ok) {
            const int kb = ks << 5;
            const unsigned short* ph = Ahi + (size_t)am * K + kb + aoff;
            const unsigned short* pl = Alo + (size_t)am * K + kb + aoff;
            ah.q[0] = *(const uint4*)ph;
            ah.q[1] = *(const uint4*)(ph + 16);
            al.q[0] = *(const uint4*)pl;
            al.q[1] = *(const uint4*)(pl + 16);
        } else {
            ah.q[0] = make_uint4(0, 0, 0, 0); ah.q[1] = make_uint4(0, 0, 0, 0);
            al.q[0] = make_uint4(0, 0, 0, 0); al.q[1] = make_uint4(0, 0, 0, 0);
        }
        const unsigned short* bh_p = bsh + (((ks * tiles_n + wave) << 9) | (lane << 4));
        const unsigned short* bl_p = bsh + ((((ksteps + ks) * tiles_n + wave) << 9) | (lane << 4));
        bh.q[0] = *(const uint4*)bh_p;
        bh.q[1] = *(const uint4*)(bh_p + 8);
        bl.q[0] = *(const uint4*)bl_p;
        bl.q[1] = *(const uint4*)(bl_p + 8);

        // hi*hi + lo*hi + hi*lo ~= fp32 product
        acc = __builtin_amdgcn_wmma_f32_16x16x32_bf16(false, ah.v, false, bh.v,
                                                      (short)0, acc, false, false);
        acc = __builtin_amdgcn_wmma_f32_16x16x32_bf16(false, al.v, false, bh.v,
                                                      (short)0, acc, false, false);
        acc = __builtin_amdgcn_wmma_f32_16x16x32_bf16(false, ah.v, false, bl.v,
                                                      (short)0, acc, false, false);
    }

    const int cn = (wave << 4) + (lane & 15);
    if (cn < N) {
        const int mbase = row0 + ((lane & 16) ? 8 : 0);
#pragma unroll
        for (int r = 0; r < 8; ++r) {
            int m = mbase + r;
            if (m < M) C[(size_t)m * N + cn] = acc[r];
        }
    }
}

// ---------------------------------------------------------------------------
// Sparse scatter kernels (edge x 4-feature parallel, fp32 atomics)
// ---------------------------------------------------------------------------
__global__ void k_spmm_scatter(const int* __restrict__ dst, const int* __restrict__ src,
                               const float* __restrict__ val, const float* __restrict__ X,
                               float* __restrict__ Y, int E, int F4, int F) {
    int t = blockIdx.x * blockDim.x + threadIdx.x;
    if (t >= E * F4) return;
    int e = t / F4, f = (t - e * F4) * 4;
    int r = dst[e], c = src[e];
    float v = val[e];
    const float4 xv = *(const float4*)(X + (size_t)c * F + f);
    float* yp = Y + (size_t)r * F + f;
    atomicAdd(yp + 0, v * xv.x);
    atomicAdd(yp + 1, v * xv.y);
    atomicAdd(yp + 2, v * xv.z);
    atomicAdd(yp + 3, v * xv.w);
}

__global__ void k_pool_scatter(const int* __restrict__ assign, const float* __restrict__ X,
                               float* __restrict__ Y, int Nsrc, int F4, int F) {
    int t = blockIdx.x * blockDim.x + threadIdx.x;
    if (t >= Nsrc * F4) return;
    int i = t / F4, f = (t - i * F4) * 4;
    int d = assign[i];
    const float4 xv = *(const float4*)(X + (size_t)i * F + f);
    float* yp = Y + (size_t)d * F + f;
    atomicAdd(yp + 0, xv.x);
    atomicAdd(yp + 1, xv.y);
    atomicAdd(yp + 2, xv.z);
    atomicAdd(yp + 3, xv.w);
}

// ---------------------------------------------------------------------------
// CRF pieces
// ---------------------------------------------------------------------------
__global__ void k_edge_logits(const int* __restrict__ dst, const int* __restrict__ src,
                              const float* __restrict__ Q, const float* __restrict__ Kt,
                              float* __restrict__ logit, unsigned* __restrict__ mOrd,
                              int E, float scale) {
    int e = blockIdx.x * blockDim.x + threadIdx.x;
    if (e >= E) return;
    const float4* q = (const float4*)(Q  + (size_t)dst[e] * NHv);
    const float4* k = (const float4*)(Kt + (size_t)src[e] * NHv);
    float s = 0.f;
#pragma unroll
    for (int j = 0; j < NHv / 4; ++j) {
        float4 a = q[j], b = k[j];
        s += a.x * b.x + a.y * b.y + a.z * b.z + a.w * b.w;
    }
    s *= scale;
    logit[e] = s;
    atomicMax(mOrd + dst[e], f2ord(s));
}

__global__ void k_edge_exp(const int* __restrict__ dst, const float* __restrict__ logit,
                           const unsigned* __restrict__ mOrd, float* __restrict__ ez,
                           float* __restrict__ zsum, int E) {
    int e = blockIdx.x * blockDim.x + threadIdx.x;
    if (e >= E) return;
    float m  = ord2f(mOrd[dst[e]]);
    float ex = expf(logit[e] - m);
    ez[e] = ex;
    atomicAdd(zsum + dst[e], ex);
}

__global__ void k_add_emb(const float* __restrict__ P, const int* __restrict__ nw,
                          const float* __restrict__ emb, float* __restrict__ h0,
                          int N, int F) {
    int t = blockIdx.x * blockDim.x + threadIdx.x;
    if (t >= N * F) return;
    int i = t / F, f = t - i * F;
    h0[t] = P[t] + emb[(size_t)nw[i] * F + f];
}

__global__ void k_crf_msg(const int* __restrict__ dst, const int* __restrict__ src,
                          const float* __restrict__ ez, const float* __restrict__ z,
                          const float* __restrict__ h0, float* __restrict__ msg,
                          int E, int F4, int F) {
    int t = blockIdx.x * blockDim.x + threadIdx.x;
    if (t >= E * F4) return;
    int e = t / F4, f = (t - e * F4) * 4;
    int d = dst[e];
    float g = ez[e] / (z[d] + 1e-16f);
    const float4 hv = *(const float4*)(h0 + (size_t)src[e] * F + f);
    float* yp = msg + (size_t)d * F + f;
    atomicAdd(yp + 0, g * hv.x);
    atomicAdd(yp + 1, g * hv.y);
    atomicAdd(yp + 2, g * hv.z);
    atomicAdd(yp + 3, g * hv.w);
}

__global__ void k_crf_combine(const float* __restrict__ h0, const float* __restrict__ msg,
                              const float* __restrict__ aP, const float* __restrict__ bP,
                              float* __restrict__ out, int n) {
    int i = blockIdx.x * blockDim.x + threadIdx.x;
    if (i >= n) return;
    float a = aP[0], b = bP[0];
    out[i] = (a * h0[i] + b * msg[i]) / (a + b);
}

// ---------------------------------------------------------------------------
// Elementwise glue
// ---------------------------------------------------------------------------
__global__ void k_bias_relu(const float* __restrict__ S, const float* __restrict__ b,
                            float* __restrict__ Y, int N, int F) {
    int t = blockIdx.x * blockDim.x + threadIdx.x;
    if (t >= N * F) return;
    float v = S[t] + b[t % F];
    Y[t] = v > 0.f ? v : 0.f;
}

__global__ void k_bias_add(float* __restrict__ Y, const float* __restrict__ b, int N, int F) {
    int t = blockIdx.x * blockDim.x + threadIdx.x;
    if (t >= N * F) return;
    Y[t] += b[t % F];
}

__global__ void k_gather_add(const int* __restrict__ assign, const float* __restrict__ Hc,
                             const float* __restrict__ skip, float* __restrict__ out,
                             int N, int F) {
    int t = blockIdx.x * blockDim.x + threadIdx.x;
    if (t >= N * F) return;
    int i = t / F, f = t - i * F;
    out[t] = Hc[(size_t)assign[i] * F + f] + skip[t];
}

// ---------------------------------------------------------------------------
// Host orchestration
// ---------------------------------------------------------------------------
extern "C" void kernel_launch(void* const* d_in, const int* in_sizes, int n_in,
                              void* d_out, int out_size, void* d_ws, size_t ws_size,
                              hipStream_t stream) {
    (void)in_sizes; (void)n_in; (void)out_size; (void)ws_size;

    const float* x      = (const float*)d_in[0];
    const int*   A0d    = (const int*)d_in[1];
    const int*   A0s    = A0d + E0v;
    const float* A0v_   = (const float*)d_in[2];
    const int*   A1d    = (const int*)d_in[3];
    const int*   A1s    = A1d + E1v;
    const int*   A2d    = (const int*)d_in[5];
    const int*   A2s    = A2d + E2v;
    const int*   assign0= (const int*)d_in[7];
    const int*   assign1= (const int*)d_in[8];
    const int*   nwgt1  = (const int*)d_in[9];
    const int*   nwgt2  = (const int*)d_in[10];
    const float* gc1_W  = (const float*)d_in[11];
    const float* gc1_b  = (const float*)d_in[12];
    const float* gc2_W  = (const float*)d_in[13];
    const float* gc2_b  = (const float*)d_in[14];
    const float* c1Wq   = (const float*)d_in[15];
    const float* c1Wk   = (const float*)d_in[16];
    const float* c1emb  = (const float*)d_in[17];
    const float* c1a    = (const float*)d_in[18];
    const float* c1b    = (const float*)d_in[19];
    const float* c2Wq   = (const float*)d_in[20];
    const float* c2Wk   = (const float*)d_in[21];
    const float* c2emb  = (const float*)d_in[22];
    const float* c2a    = (const float*)d_in[23];
    const float* c2b    = (const float*)d_in[24];

    float* out  = (float*)d_out;                    // [N0, 40]
    float* gcnh = out  + (size_t)N0v * NCv;         // [N0, 128]
    float* crfh = gcnh + (size_t)N0v * NHv;         // [N0, 128]

    // bump allocator over workspace (256B aligned regions)
    char* wsp = (char*)d_ws;
    auto alloc = [&](size_t bytes) -> void* {
        void* p = (void*)wsp;
        wsp += (bytes + 255) & ~(size_t)255;
        return p;
    };
    auto g1 = [](size_t n) { return (unsigned)((n + 255) / 256); };

    auto split = [&](const float* s, unsigned short* hi, unsigned short* lo, size_t n) {
        k_split_bf16<<<g1(n), 256, 0, stream>>>(s, hi, lo, (int)n);
    };
    auto zfill = [&](float* p, size_t n) {
        k_zfill<<<g1(n), 256, 0, stream>>>(p, (int)n);
    };
    // pack weights to fragment order + run tiled WMMA GEMM
    auto gemm = [&](const unsigned short* Ah, const unsigned short* Al,
                    const float* W, float* Cm, int M, int N, int K) {
        int tiles_n = (N + 15) / 16;
        int total = 2 * (K / 32) * tiles_n * 512;            // packed halfs
        unsigned short* Bpk = (unsigned short*)alloc((size_t)total * 2);
        k_pack_b<<<g1((size_t)total), 256, 0, stream>>>(W, Bpk, N, K);
        k_wmma_gemm_bf16x3<<<(M + 15) / 16, 256, 0, stream>>>(Ah, Al, Bpk, Cm, M, N, K);
    };

    // CRF refinement for one level
    auto run_crf = [&](const float* P, int n, int Ee,
                       const int* dstI, const int* srcI, const int* nw,
                       const float* Wq, const float* Wk, const float* emb,
                       const float* aP, const float* bP, float* Cout) {
        size_t nf = (size_t)n * NHv;
        unsigned short* phi = (unsigned short*)alloc(nf * 2);
        unsigned short* plo = (unsigned short*)alloc(nf * 2);
        float*    q   = (float*)alloc(nf * 4);
        float*    kk  = (float*)alloc(nf * 4);
        float*    lg  = (float*)alloc((size_t)Ee * 4);
        float*    ez  = (float*)alloc((size_t)Ee * 4);
        unsigned* mOrd= (unsigned*)alloc((size_t)n * 4);
        float*    z   = (float*)alloc((size_t)n * 4);
        float*    h0  = (float*)alloc(nf * 4);
        float*    msg = (float*)alloc(nf * 4);

        split(P, phi, plo, nf);
        gemm(phi, plo, Wq, q,  n, NHv, NHv);
        gemm(phi, plo, Wk, kk, n, NHv, NHv);
        zfill((float*)mOrd, (size_t)n);   // 0 == ordered "-NaN": below every real
        zfill(z, (size_t)n);
        const float scale = 0.08838834764831845f;   // 1/sqrt(128)
        k_edge_logits<<<g1((size_t)Ee), 256, 0, stream>>>(dstI, srcI, q, kk, lg, mOrd, Ee, scale);
        k_edge_exp<<<g1((size_t)Ee), 256, 0, stream>>>(dstI, lg, mOrd, ez, z, Ee);
        k_add_emb<<<g1(nf), 256, 0, stream>>>(P, nw, emb, h0, n, NHv);
        zfill(msg, nf);
        k_crf_msg<<<g1((size_t)Ee * (NHv / 4)), 256, 0, stream>>>(dstI, srcI, ez, z, h0, msg,
                                                                  Ee, NHv / 4, NHv);
        k_crf_combine<<<g1(nf), 256, 0, stream>>>(h0, msg, aP, bP, Cout, (int)nf);
    };

    // ---------------- Stage 1: gc1 + relu -> gcn_hidden ----------------
    unsigned short* xhi = (unsigned short*)alloc((size_t)N0v * NFv * 2);
    unsigned short* xlo = (unsigned short*)alloc((size_t)N0v * NFv * 2);
    float* xw1 = (float*)alloc((size_t)N0v * NHv * 4);
    float* S1  = (float*)alloc((size_t)N0v * NHv * 4);

    split(x, xhi, xlo, (size_t)N0v * NFv);
    gemm(xhi, xlo, gc1_W, xw1, N0v, NHv, NFv);
    zfill(S1, (size_t)N0v * NHv);
    k_spmm_scatter<<<g1((size_t)E0v * (NHv / 4)), 256, 0, stream>>>(A0d, A0s, A0v_, xw1, S1,
                                                                    E0v, NHv / 4, NHv);
    k_bias_relu<<<g1((size_t)N0v * NHv), 256, 0, stream>>>(S1, gc1_b, gcnh, N0v, NHv);

    // ---------------- Level 1: pool + CRF ----------------
    float* p1 = (float*)alloc((size_t)N1v * NHv * 4);
    float* c1 = (float*)alloc((size_t)N1v * NHv * 4);
    zfill(p1, (size_t)N1v * NHv);
    k_pool_scatter<<<g1((size_t)N0v * (NHv / 4)), 256, 0, stream>>>(assign0, gcnh, p1,
                                                                    N0v, NHv / 4, NHv);
    run_crf(p1, N1v, E1v, A1d, A1s, nwgt1, c1Wq, c1Wk, c1emb, c1a, c1b, c1);

    // ---------------- Level 2: pool + CRF ----------------
    float* p2 = (float*)alloc((size_t)N2v * NHv * 4);
    float* c2 = (float*)alloc((size_t)N2v * NHv * 4);
    zfill(p2, (size_t)N2v * NHv);
    k_pool_scatter<<<g1((size_t)N1v * (NHv / 4)), 256, 0, stream>>>(assign1, c1, p2,
                                                                    N1v, NHv / 4, NHv);
    run_crf(p2, N2v, E2v, A2d, A2s, nwgt2, c2Wq, c2Wk, c2emb, c2a, c2b, c2);

    // ---------------- Unpool with skip connections -> crf_hidden ----------------
    float* u1 = (float*)alloc((size_t)N1v * NHv * 4);
    k_gather_add<<<g1((size_t)N1v * NHv), 256, 0, stream>>>(assign1, c2, c1, u1, N1v, NHv);
    k_gather_add<<<g1((size_t)N0v * NHv), 256, 0, stream>>>(assign0, u1, gcnh, crfh, N0v, NHv);

    // ---------------- Stage 2: gc2 -> out ----------------
    unsigned short* chh = (unsigned short*)alloc((size_t)N0v * NHv * 2);
    unsigned short* chl = (unsigned short*)alloc((size_t)N0v * NHv * 2);
    float* cw2 = (float*)alloc((size_t)N0v * NCv * 4);

    split(crfh, chh, chl, (size_t)N0v * NHv);
    gemm(chh, chl, gc2_W, cw2, N0v, NCv, NHv);
    zfill(out, (size_t)N0v * NCv);
    k_spmm_scatter<<<g1((size_t)E0v * (NCv / 4)), 256, 0, stream>>>(A0d, A0s, A0v_, cw2, out,
                                                                    E0v, NCv / 4, NCv);
    k_bias_add<<<g1((size_t)N0v * NCv), 256, 0, stream>>>(out, gc2_b, N0v, NCv);
}
